// DecLayerJ_38208029065283
// MI455X (gfx1250) — matrix operand order repaired and checked by reading, not compiled, not run on hardware
//
#include <hip/hip_runtime.h>

#define H 128
#define CE 384
#define KNB 48
#define NNODES 8192   // B*N = 2*4096
#define SCALE 30.0f

typedef __attribute__((ext_vector_type(16))) _Float16 v16h;
typedef __attribute__((ext_vector_type(8)))  float    v8f;

union HF   { _Float16 h; unsigned short u; };
union Frag { v16h v; unsigned int u[8]; uint4 q[2]; };

// CDNA5 has V_TANH_F32 (hardware transcendental); fall back to exp-based tanh.
__device__ inline float fast_tanh_f(float x){
#if __has_builtin(__builtin_amdgcn_tanhf)
  return __builtin_amdgcn_tanhf(x);
#else
  float e = __expf(2.0f * x);          // v_exp_f32
  return 1.0f - 2.0f / (e + 1.0f);
#endif
}

__device__ inline float gelu_tanh_f(float x){
  float x3 = x*x*x;
  return 0.5f*x*(1.0f + fast_tanh_f(0.7978845608028654f*(x + 0.044715f*x3)));
}
__device__ inline float gelu_erf_f(float x){
  return 0.5f*x*(1.0f + erff(x*0.7071067811865475f));
}

// A fragment: 16x32 f16 from row-major LDS (stride in halfs). ISA 7.12.2 layout:
// lane m=lane&15 is row M; hf=lane>>4 selects K 0-7/16-23 vs 8-15/24-31.
// Per lane the 16 halfs are two contiguous 16B runs -> 2x ds_load_b128.
__device__ inline v16h load_a16(const _Float16* base, int stride, int r0, int k0, int lane){
  const int m = lane & 15, hf = lane >> 4;
  const _Float16* p = base + (r0 + m)*stride + k0 + (hf << 3);
  Frag f;
  f.q[0] = *(const uint4*)p;          // k = hf*8 + 0..7
  f.q[1] = *(const uint4*)(p + 16);   // k = 16 + hf*8 + 0..7
  return f.v;
}

// B fragment: 32x16 f16 from fragment-major packed weights:
// Wf[((kt*nTiles + nt)*32 + lane)*16 + j]; lane&15 = column, lane>>4 selects K 0-7/16-23 vs 8-15/24-31.
// Each lane's fragment is one contiguous 32B run -> 2x ds_load_b128.
__device__ inline v16h load_bfrag(const _Float16* wp, int nTiles, int kt, int nt, int lane){
  const _Float16* p = wp + ((((kt*nTiles + nt) << 5) + lane) << 4);
  Frag f;
  f.q[0] = *(const uint4*)p;
  f.q[1] = *(const uint4*)(p + 8);
  return f.v;
}

__device__ inline v8f wmma_f16(v16h a, v16h b, v8f c){
  return __builtin_amdgcn_wmma_f32_16x16x32_f16(false, a, false, b, (short)0, c, false, false);
}

// -------- prep: f32 weights -> f16 fragment-major packed --------
__global__ void pack_weights_kernel(const float* __restrict__ w, _Float16* __restrict__ dst,
                                    int KK, int NN){
  int i = blockIdx.x*blockDim.x + threadIdx.x;
  if (i >= KK*NN) return;
  int k = i / NN, n = i - k*NN;
  int kt = k >> 5, kr = k & 31, hf = (kr >> 3) & 1, j = (kr & 7) | ((kr >> 4) << 3);
  int nt = n >> 4, lane = (hf << 4) | (n & 15);
  dst[(((kt*(NN >> 4) + nt)*32 + lane) << 4) + j] = (_Float16)w[i];
}

// -------- kernel 1: per-node edge MLP + masked K-aggregation --------
__global__ __launch_bounds__(256) void edge_mlp_kernel(
    const float* __restrict__ hV, const float* __restrict__ hE,
    const float* __restrict__ maskA_g,
    const _Float16* __restrict__ Wp,      // packed W1|W2|W3
    const float* __restrict__ b1, const float* __restrict__ b2, const float* __restrict__ b3,
    float* __restrict__ hmid)
{
  extern __shared__ char smem[];
  _Float16* A  = (_Float16*)(smem);           // 49152 B  (48x512)
  _Float16* W1 = (_Float16*)(smem + 49152);   // 131072 B
  _Float16* W2 = (_Float16*)(smem + 180224);  // 32768 B
  _Float16* W3 = (_Float16*)(smem + 212992);  // 32768 B
  _Float16* X1 = (_Float16*)(smem + 245760);  // 12288 B  (48x128)
  _Float16* X2 = (_Float16*)(smem + 258048);  // 12288 B
  float*    MA = (float*)   (smem + 270336);  // 192 B

  const int node = blockIdx.x;
  const int tid  = threadIdx.x;
  const int lane = tid & 31;
  const int wv   = tid >> 5;
  const int col  = (wv << 4) + (lane & 15);
  const int hf   = lane >> 4;

  // stage packed weights (contiguous 196608 B) into LDS as b128s
  {
    const uint4* src = (const uint4*)Wp;
    uint4* dst = (uint4*)W1;
    for (int i = tid; i < 12288; i += 256) dst[i] = src[i];
  }
  // build A = [h_V | h_E] in f16, pair stores
  for (int p = tid; p < KNB*256; p += 256){
    int k = p >> 8, c = (p & 255) << 1;
    float2 v2;
    if (c < H) v2 = *(const float2*)(hV + (size_t)node*H + c);
    else       v2 = *(const float2*)(hE + ((size_t)node*KNB + k)*CE + (c - H));
    HF a0, a1; a0.h = (_Float16)v2.x; a1.h = (_Float16)v2.y;
    *(unsigned int*)(A + k*512 + c) = (unsigned int)a0.u | ((unsigned int)a1.u << 16);
  }
  if (tid < KNB) MA[tid] = maskA_g[(size_t)node*KNB + tid];
  __syncthreads();

  // ---- layer 1: [48x512] @ [512x128], gelu_tanh ----
  {
    const float bias = b1[col];
    v8f c0 = {}, c1 = {}, c2 = {};
#pragma unroll
    for (int ks = 0; ks < 16; ++ks){
      v16h b = load_bfrag(W1, 8, ks, wv, lane);           // shared across M-tiles
      c0 = wmma_f16(load_a16(A, 512,  0, ks*32, lane), b, c0);
      c1 = wmma_f16(load_a16(A, 512, 16, ks*32, lane), b, c1);
      c2 = wmma_f16(load_a16(A, 512, 32, ks*32, lane), b, c2);
    }
#pragma unroll
    for (int g = 0; g < 8; ++g){
      int r = g + (hf << 3);
      X1[(r +  0)*H + col] = (_Float16)gelu_tanh_f(c0[g] + bias);
      X1[(r + 16)*H + col] = (_Float16)gelu_tanh_f(c1[g] + bias);
      X1[(r + 32)*H + col] = (_Float16)gelu_tanh_f(c2[g] + bias);
    }
  }
  __syncthreads();

  // ---- layer 2: [48x128] @ [128x128], gelu_tanh ----
  {
    const float bias = b2[col];
    v8f c0 = {}, c1 = {}, c2 = {};
#pragma unroll
    for (int ks = 0; ks < 4; ++ks){
      v16h b = load_bfrag(W2, 8, ks, wv, lane);
      c0 = wmma_f16(load_a16(X1, H,  0, ks*32, lane), b, c0);
      c1 = wmma_f16(load_a16(X1, H, 16, ks*32, lane), b, c1);
      c2 = wmma_f16(load_a16(X1, H, 32, ks*32, lane), b, c2);
    }
#pragma unroll
    for (int g = 0; g < 8; ++g){
      int r = g + (hf << 3);
      X2[(r +  0)*H + col] = (_Float16)gelu_tanh_f(c0[g] + bias);
      X2[(r + 16)*H + col] = (_Float16)gelu_tanh_f(c1[g] + bias);
      X2[(r + 32)*H + col] = (_Float16)gelu_tanh_f(c2[g] + bias);
    }
  }
  __syncthreads();

  // ---- layer 3 + masked sum over K / SCALE + residual ----
  {
    const float bias = b3[col];
    v8f c0 = {}, c1 = {}, c2 = {};
#pragma unroll
    for (int ks = 0; ks < 4; ++ks){
      v16h b = load_bfrag(W3, 8, ks, wv, lane);
      c0 = wmma_f16(load_a16(X2, H,  0, ks*32, lane), b, c0);
      c1 = wmma_f16(load_a16(X2, H, 16, ks*32, lane), b, c1);
      c2 = wmma_f16(load_a16(X2, H, 32, ks*32, lane), b, c2);
    }
    float partial = 0.0f;
#pragma unroll
    for (int g = 0; g < 8; ++g){
      int r = g + (hf << 3);
      partial += MA[r +  0] * (c0[g] + bias);
      partial += MA[r + 16] * (c1[g] + bias);
      partial += MA[r + 32] * (c2[g] + bias);
    }
    partial += __shfl_xor(partial, 16, 32);   // combine lane halves (rows)
    if (lane < 16){
      hmid[(size_t)node*H + col] = hV[(size_t)node*H + col] + partial*(1.0f/SCALE);
    }
  }
}

// -------- kernel 2: FFN over [B*N, H], 32 rows per workgroup --------
__global__ __launch_bounds__(256) void ffn_kernel(
    const float* __restrict__ hmid, const float* __restrict__ maskV,
    const _Float16* __restrict__ Wffp,    // packed Win|Wout
    const float* __restrict__ bin, const float* __restrict__ bout,
    float* __restrict__ out)
{
  extern __shared__ char smem[];
  _Float16* A  = (_Float16*)(smem);            // 8192 B  (32x128)
  _Float16* Wi = (_Float16*)(smem + 8192);     // 131072 B
  _Float16* Wo = (_Float16*)(smem + 139264);   // 131072 B
  _Float16* T  = (_Float16*)(smem + 270336);   // 32768 B (32x512)

  const int r0  = blockIdx.x << 5;
  const int tid = threadIdx.x, lane = tid & 31, wv = tid >> 5, hf = lane >> 4;

  {
    const uint4* src = (const uint4*)Wffp;
    uint4* dst = (uint4*)Wi;
    for (int i = tid; i < 16384; i += 256) dst[i] = src[i];
  }
  for (int p = tid; p < 32*64; p += 256){
    int r = p >> 6, c = (p & 63) << 1;
    float2 v2 = *(const float2*)(hmid + (size_t)(r0 + r)*H + c);
    HF a0, a1; a0.h = (_Float16)v2.x; a1.h = (_Float16)v2.y;
    *(unsigned int*)(A + r*H + c) = (unsigned int)a0.u | ((unsigned int)a1.u << 16);
  }
  __syncthreads();

  // ---- GEMM1: [32x128] @ [128x512], gelu_exact ----
#pragma unroll
  for (int j = 0; j < 4; ++j){
    const int nt = wv + (j << 3);
    const int ccol = (nt << 4) + (lane & 15);
    const float bias = bin[ccol];
    v8f c0 = {}, c1 = {};
#pragma unroll
    for (int ks = 0; ks < 4; ++ks){
      v16h b = load_bfrag(Wi, 32, ks, nt, lane);
      c0 = wmma_f16(load_a16(A, H,  0, ks*32, lane), b, c0);
      c1 = wmma_f16(load_a16(A, H, 16, ks*32, lane), b, c1);
    }
#pragma unroll
    for (int g = 0; g < 8; ++g){
      int r = g + (hf << 3);
      T[(r +  0)*512 + ccol] = (_Float16)gelu_erf_f(c0[g] + bias);
      T[(r + 16)*512 + ccol] = (_Float16)gelu_erf_f(c1[g] + bias);
    }
  }
  __syncthreads();

  // ---- GEMM2: [32x512] @ [512x128] + residual + mask_V ----
  {
    const int ccol = (wv << 4) + (lane & 15);
    const float bo = bout[ccol];
    v8f c0 = {}, c1 = {};
#pragma unroll
    for (int ks = 0; ks < 16; ++ks){
      v16h b = load_bfrag(Wo, 8, ks, wv, lane);
      c0 = wmma_f16(load_a16(T, 512,  0, ks*32, lane), b, c0);
      c1 = wmma_f16(load_a16(T, 512, 16, ks*32, lane), b, c1);
    }
#pragma unroll
    for (int g = 0; g < 8; ++g){
      int r = g + (hf << 3);
      int row0 = r0 + r, row1 = r0 + r + 16;
      out[(size_t)row0*H + ccol] = maskV[row0] * (hmid[(size_t)row0*H + ccol] + c0[g] + bo);
      out[(size_t)row1*H + ccol] = maskV[row1] * (hmid[(size_t)row1*H + ccol] + c1[g] + bo);
    }
  }
}

extern "C" void kernel_launch(void* const* d_in, const int* in_sizes, int n_in,
                              void* d_out, int out_size, void* d_ws, size_t ws_size,
                              hipStream_t stream){
  const float* hV    = (const float*)d_in[0];
  const float* hE    = (const float*)d_in[1];
  const float* maskV = (const float*)d_in[2];
  const float* maskA = (const float*)d_in[3];
  const float* W1w   = (const float*)d_in[4];
  const float* W1b   = (const float*)d_in[5];
  const float* W2w   = (const float*)d_in[6];
  const float* W2b   = (const float*)d_in[7];
  const float* W3w   = (const float*)d_in[8];
  const float* W3b   = (const float*)d_in[9];
  const float* Winw  = (const float*)d_in[10];
  const float* Winb  = (const float*)d_in[11];
  const float* Woutw = (const float*)d_in[12];
  const float* Woutb = (const float*)d_in[13];

  char* ws = (char*)d_ws;
  _Float16* W1p   = (_Float16*)(ws + 0);        // 131072 B
  _Float16* W2p   = (_Float16*)(ws + 131072);   // 32768 B
  _Float16* W3p   = (_Float16*)(ws + 163840);   // 32768 B
  _Float16* Winp  = (_Float16*)(ws + 196608);   // 131072 B
  _Float16* Woutp = (_Float16*)(ws + 327680);   // 131072 B
  float*    hmid  = (float*)   (ws + 458752);   // 4 MB

  pack_weights_kernel<<<(512*128 + 255)/256, 256, 0, stream>>>(W1w,   W1p,   512, 128);
  pack_weights_kernel<<<(128*128 + 255)/256, 256, 0, stream>>>(W2w,   W2p,   128, 128);
  pack_weights_kernel<<<(128*128 + 255)/256, 256, 0, stream>>>(W3w,   W3p,   128, 128);
  pack_weights_kernel<<<(128*512 + 255)/256, 256, 0, stream>>>(Winw,  Winp,  128, 512);
  pack_weights_kernel<<<(512*128 + 255)/256, 256, 0, stream>>>(Woutw, Woutp, 512, 128);

  edge_mlp_kernel<<<NNODES, 256, 270528, stream>>>(hV, hE, maskA, W1p, W1b, W2b, W3b, hmid);
  ffn_kernel<<<NNODES/32, 256, 303104, stream>>>(hmid, maskV, Winp, Winb, Woutb, (float*)d_out);
}
